// KMeansPT_55671366091569
// MI455X (gfx1250) — matrix operand order repaired and checked by reading, not compiled, not run on hardware
//
#include <hip/hip_runtime.h>
#include <math.h>

typedef __attribute__((ext_vector_type(16))) __bf16 v16bf;
typedef __attribute__((ext_vector_type(8)))  __bf16 v8bf;
typedef __attribute__((ext_vector_type(8)))  float  v8f;
typedef __attribute__((ext_vector_type(4)))  float  v4f;

#define N_TOTAL 500000
#define DIM     128
#define KC      256
#define M_TILE  32
#define DPAD    257   // padded LDS row stride (floats)

// ---------------------------------------------------------------------------
// CDNA5 async global->LDS copy (ASYNCcnt path), ISA §15.18.3 op 98.
// VDST = per-lane LDS byte address (low 32 bits of generic shared pointer,
// per the flat-aperture rule); VADDR = global address.
// ---------------------------------------------------------------------------
__device__ __forceinline__ void async_load16(unsigned lds_off, const float* g) {
    asm volatile("global_load_async_to_lds_b128 %0, %1, off"
                 :: "v"(lds_off), "v"(g) : "memory");
}
__device__ __forceinline__ void wait_async0() {
    asm volatile("s_wait_asynccnt 0x0" ::: "memory");
}

// A fragment (16x32 bf16) from bf16 LDS row, contiguous K layout:
// elems 0..15 -> K = kofs + e
__device__ __forceinline__ v16bf load_a_bf(const __bf16* rowp, int kofs) {
    union { v16bf v; v8bf h[2]; } a;
    a.h[0] = *(const v8bf*)(rowp + kofs);
    a.h[1] = *(const v8bf*)(rowp + kofs + 8);
    return a.v;
}

// B fragment: pre-swizzled bf16 V copy, 32 contiguous bytes per lane.
__device__ __forceinline__ v16bf load_b_frag(const __bf16* __restrict__ Vbf,
                                             int frag, int lane) {
    const __bf16* p = Vbf + ((long)frag * 32 + lane) * 16;
    union { v16bf v; v8bf h[2]; } b;
    b.h[0] = *(const v8bf*)(p);
    b.h[1] = *(const v8bf*)(p + 8);
    return b.v;
}

// ---------------------------------------------------------------------------
// Prep: per-centroid squared norms + pre-swizzled bf16 copy of V.
// Fragment (w,s,t): lane l, elem e -> V[w*32 + t*16 + (l&15)][s*32 + (l>>4)*16 + e]
// ---------------------------------------------------------------------------
__global__ void __launch_bounds__(256)
prep_kernel(const float* __restrict__ V, float* __restrict__ vsq,
            __bf16* __restrict__ Vbf) {
    const int tid = threadIdx.x;
    if (blockIdx.x == 0) {                       // vsq: one thread per centroid
        const float* row = V + (long)tid * DIM;
        float s = 0.f;
#pragma unroll
        for (int i = 0; i < DIM; i += 4) {
            v4f x = *(const v4f*)(row + i);
            s += x.x*x.x + x.y*x.y + x.z*x.z + x.w*x.w;
        }
        vsq[tid] = s;
    }
    const int j    = blockIdx.x * 256 + tid;     // 0..2047 = 64 frags x 32 lanes
    const int frag = j >> 5;
    const int lane = j & 31;
    const int w = frag >> 3, s = (frag >> 1) & 3, t = frag & 1;
    const int kc = w * 32 + t * 16 + (lane & 15);
    const int d0 = s * 32 + (lane >> 4) * 16;
    const float* src = V + (long)kc * DIM + d0;
    __bf16* dst = Vbf + ((long)frag * 32 + lane) * 16;
#pragma unroll
    for (int i = 0; i < 16; i += 4) {
        v4f f = *(const v4f*)(src + i);
        dst[i+0] = (__bf16)f.x; dst[i+1] = (__bf16)f.y;
        dst[i+2] = (__bf16)f.z; dst[i+3] = (__bf16)f.w;
    }
}

// ---------------------------------------------------------------------------
// Main: 32-row tile per workgroup; 8 waves, each owning 32 centroid columns
// over both 16-row subtiles -> 16 v_wmma_f32_16x16x32_bf16 per wave.
// ---------------------------------------------------------------------------
__global__ void __launch_bounds__(256)
kmeans_tile_kernel(const float* __restrict__ X, const __bf16* __restrict__ Vbf,
                   const float* __restrict__ vsq_g, float* __restrict__ out) {
    __shared__ __align__(16) float  Xf[M_TILE][DIM];   // 16 KB, async-staged f32
    __shared__ __align__(16) __bf16 Xs[M_TILE][DIM];   //  8 KB, bf16 copy
    __shared__ float vsq[KC];
    __shared__ float xsq[M_TILE];
    __shared__ float dist[M_TILE][DPAD];               // ~32.1 KB
    __shared__ float minv[M_TILE][8];
    __shared__ int   mini[M_TILE][8];
    __shared__ int   argidx[M_TILE];

    const int tid  = threadIdx.x;
    const int wave = tid >> 5;
    const int lane = tid & 31;
    const int half = lane >> 4;
    const int nlo  = lane & 15;
    const long n0  = (long)blockIdx.x * M_TILE;

    // ---- async-stage the X tile (32x128 f32 = 16 KB) into LDS ----
    {
        const int row = tid >> 3;            // 0..31
        const int cb  = (tid & 7) * 16;      // 16 consecutive floats per thread
        const float* src = X + (n0 + row) * DIM + cb;
        unsigned lds = (unsigned)(uintptr_t)&Xf[row][cb];
        async_load16(lds,      src);
        async_load16(lds + 16, src + 4);
        async_load16(lds + 32, src + 8);
        async_load16(lds + 48, src + 12);
    }
    vsq[tid] = vsq_g[tid];
    if (tid < M_TILE) xsq[tid] = 0.f;
    wait_async0();
    __syncthreads();

    // ---- single pass: fp32 row norms (ds_add_f32) + bf16 conversion ----
    {
        const int row = tid >> 3;
        const int cb  = (tid & 7) * 16;
        v4f p0 = *(const v4f*)&Xf[row][cb];
        v4f p1 = *(const v4f*)&Xf[row][cb + 4];
        v4f p2 = *(const v4f*)&Xf[row][cb + 8];
        v4f p3 = *(const v4f*)&Xf[row][cb + 12];
        float s = p0.x*p0.x + p0.y*p0.y + p0.z*p0.z + p0.w*p0.w
                + p1.x*p1.x + p1.y*p1.y + p1.z*p1.z + p1.w*p1.w
                + p2.x*p2.x + p2.y*p2.y + p2.z*p2.z + p2.w*p2.w
                + p3.x*p3.x + p3.y*p3.y + p3.z*p3.z + p3.w*p3.w;
        v8bf h0, h1;
        h0[0]=(__bf16)p0.x; h0[1]=(__bf16)p0.y; h0[2]=(__bf16)p0.z; h0[3]=(__bf16)p0.w;
        h0[4]=(__bf16)p1.x; h0[5]=(__bf16)p1.y; h0[6]=(__bf16)p1.z; h0[7]=(__bf16)p1.w;
        h1[0]=(__bf16)p2.x; h1[1]=(__bf16)p2.y; h1[2]=(__bf16)p2.z; h1[3]=(__bf16)p2.w;
        h1[4]=(__bf16)p3.x; h1[5]=(__bf16)p3.y; h1[6]=(__bf16)p3.z; h1[7]=(__bf16)p3.w;
        *(v8bf*)&Xs[row][cb]     = h0;
        *(v8bf*)&Xs[row][cb + 8] = h1;
        atomicAdd(&xsq[row], s);
    }
    __syncthreads();

    // ---- WMMA main loop: 16 bf16 WMMAs per wave, no per-wave conversions ----
    v8f acc00 = {}, acc01 = {}, acc10 = {}, acc11 = {};
#pragma unroll
    for (int s = 0; s < DIM / 32; ++s) {
        const int kofs = s * 32 + half * 16;
        v16bf a0 = load_a_bf(&Xs[nlo][0],      kofs);   // rows 0..15
        v16bf a1 = load_a_bf(&Xs[nlo + 16][0], kofs);   // rows 16..31
        v16bf b0 = load_b_frag(Vbf, (wave * 4 + s) * 2 + 0, lane);
        acc00 = __builtin_amdgcn_wmma_f32_16x16x32_bf16(false, a0, false, b0,
                                                        (short)0, acc00, false, false);
        acc10 = __builtin_amdgcn_wmma_f32_16x16x32_bf16(false, a1, false, b0,
                                                        (short)0, acc10, false, false);
        v16bf b1 = load_b_frag(Vbf, (wave * 4 + s) * 2 + 1, lane);
        acc01 = __builtin_amdgcn_wmma_f32_16x16x32_bf16(false, a0, false, b1,
                                                        (short)0, acc01, false, false);
        acc11 = __builtin_amdgcn_wmma_f32_16x16x32_bf16(false, a1, false, b1,
                                                        (short)0, acc11, false, false);
    }

    // ---- dist = sqrt(max(x2 + v2 - 2*dot, 0)) into LDS ----
    const int k0 = wave * 32 + nlo;
    const int k1 = k0 + 16;
    const float vs0 = vsq[k0];
    const float vs1 = vsq[k1];
#pragma unroll
    for (int r = 0; r < 8; ++r) {
        const int row = r + half * 8;              // C/D layout: M = r + half*8
        const float xa = xsq[row];
        const float xb = xsq[row + 16];
        dist[row][k0]      = sqrtf(fmaxf(xa + vs0 - 2.f * acc00[r], 0.f));
        dist[row][k1]      = sqrtf(fmaxf(xa + vs1 - 2.f * acc01[r], 0.f));
        dist[row + 16][k0] = sqrtf(fmaxf(xb + vs0 - 2.f * acc10[r], 0.f));
        dist[row + 16][k1] = sqrtf(fmaxf(xb + vs1 - 2.f * acc11[r], 0.f));
    }
    __syncthreads();

    // ---- argmin stage 1: 8 threads/row, 32 cols each (first-index tiebreak) ----
    {
        const int row = tid >> 3;
        const int seg = tid & 7;
        const int cb  = seg * 32;
        float mv = dist[row][cb];
        int   mi = cb;
#pragma unroll
        for (int c = 1; c < 32; ++c) {
            const float v = dist[row][cb + c];
            if (v < mv) { mv = v; mi = cb + c; }
        }
        minv[row][seg] = mv;
        mini[row][seg] = mi;
    }
    __syncthreads();

    // ---- argmin stage 2: one thread per row ----
    if (tid < M_TILE) {
        float mv = minv[tid][0];
        int   mi = mini[tid][0];
#pragma unroll
        for (int s2 = 1; s2 < 8; ++s2) {
            const float v = minv[tid][s2];
            const int   i = mini[tid][s2];
            if (v < mv || (v == mv && i < mi)) { mv = v; mi = i; }
        }
        argidx[tid] = mi;
    }
    __syncthreads();

    // ---- write one-hot-masked tile, non-temporal, 128 B per thread ----
    {
        const int row = tid >> 3;
        const int cb  = (tid & 7) * 32;
        const int am  = argidx[row];
        const float dm = dist[row][am];
        float* dst = out + (n0 + row) * KC + cb;
#pragma unroll
        for (int c = 0; c < 32; c += 4) {
            v4f v;
            v.x = (cb + c + 0 == am) ? dm : 0.f;
            v.y = (cb + c + 1 == am) ? dm : 0.f;
            v.z = (cb + c + 2 == am) ? dm : 0.f;
            v.w = (cb + c + 3 == am) ? dm : 0.f;
            __builtin_nontemporal_store(v, (v4f*)(dst + c));
        }
    }
}

// ---------------------------------------------------------------------------
extern "C" void kernel_launch(void* const* d_in, const int* in_sizes, int n_in,
                              void* d_out, int out_size, void* d_ws, size_t ws_size,
                              hipStream_t stream) {
    (void)in_sizes; (void)n_in; (void)out_size; (void)ws_size;
    const float* X = (const float*)d_in[0];     // [N, 128]
    const float* V = (const float*)d_in[1];     // [256, 128]
    float* out = (float*)d_out;                 // [N, 256]

    float*  vsq = (float*)d_ws;                             // 1 KB
    __bf16* Vbf = (__bf16*)((char*)d_ws + 1024);            // 64 KB swizzled V

    prep_kernel<<<8, 256, 0, stream>>>(V, vsq, Vbf);

    const int nblocks = N_TOTAL / M_TILE;       // 15625, exact
    kmeans_tile_kernel<<<nblocks, 256, 0, stream>>>(X, Vbf, vsq, out);
}
